// ScaledDotProductAttention_16939351016164
// MI455X (gfx1250) — compile-verified, matrix operand order checked by reading, not run on hardware
//
#include <hip/hip_runtime.h>

typedef __attribute__((ext_vector_type(16))) _Float16 v16h;
typedef __attribute__((ext_vector_type(8)))  _Float16 v8h;
typedef __attribute__((ext_vector_type(8)))  float    v8f;
typedef __attribute__((ext_vector_type(4)))  float    v4f;

#define NW  2048
#define DK  64
#define DV  64
#define NBH 32            // NB*H
#define QT  (NW / 16)     // 128 query tiles per (b,h)
#define KB32 (NW / 32)    // 64 key blocks of 32
#define WAVES_PER_BLOCK 8
#define NQ  ((size_t)NBH * NW * DK)   // elements in Q (= K = V = mask): 4,194,304
#define OSTRIDE 68                    // O-stage row stride in floats (272B, 16B-aligned)
#define L2E 1.44269504088896f         // log2(e)

// 16-bit A/B operand layout (ISA 7.12.2): lane half h, v16h element e maps to
// contraction index K = e + 8*h + (e>=8 ? 8 : 0)  ==> two contiguous 8-elem runs
// starting at K = 8h and K = 16+8h. Load them as two 16-byte vectors.
__device__ __forceinline__ v16h load_ab(const _Float16* __restrict__ p, int h) {
    v8h lo = *(const v8h*)(p + 8 * h);
    v8h hi = *(const v8h*)(p + 16 + 8 * h);
    return __builtin_shufflevector(lo, hi, 0, 1, 2, 3, 4, 5, 6, 7,
                                           8, 9, 10, 11, 12, 13, 14, 15);
}

__device__ __forceinline__ float tree_max8(const v8f& v) {
    float a = fmaxf(v[0], v[1]), b = fmaxf(v[2], v[3]);
    float c = fmaxf(v[4], v[5]), d = fmaxf(v[6], v[7]);
    return fmaxf(fmaxf(a, b), fmaxf(c, d));
}
__device__ __forceinline__ float tree_sum8(const v8f& v) {
    return ((v[0] + v[1]) + (v[2] + v[3])) + ((v[4] + v[5]) + (v[6] + v[7]));
}
// exp(x - m) = exp2(fma(x, log2e, nm2)) with nm2 = -m*log2e : one fma + v_exp_f32
__device__ __forceinline__ float exp_fma(float x, float nm2) {
    return __builtin_amdgcn_exp2f(fmaf(x, L2E, nm2));
}

// ---------------- pass 0: one-time operand preparation ----------------
// Qh: f16 row-major, PRE-SCALED by 1/sqrt(dk)=0.125 so WMMA emits final scores.
// Kh: f16 row-major. Vt: f16 transposed [bh][dv][key].
// biasT / biasR: mask bias (-1e9 / 0) in tile-major order for pass 1 (S^T) and
// pass 2 (S) respectively, so each lane's C-operand init is contiguous b128s.
__global__ __launch_bounds__(256) void prep_kernel(
    const float* __restrict__ Q, const float* __restrict__ K,
    const float* __restrict__ V, const int* __restrict__ mask,
    _Float16* __restrict__ Qh, _Float16* __restrict__ Kh,
    _Float16* __restrict__ Vt, float* __restrict__ biasT,
    float* __restrict__ biasR)
{
    const size_t i = (size_t)blockIdx.x * blockDim.x + threadIdx.x;
    if (i >= NQ) return;
    Qh[i] = (_Float16)(Q[i] * 0.125f);
    Kh[i] = (_Float16)K[i];
    // transpose V within each (b,h): [key][dv] -> [dv][key]
    {
        const size_t bh  = i / ((size_t)NW * DV);
        const size_t rem = i % ((size_t)NW * DV);
        const size_t key = rem / DV;
        const size_t dv  = rem % DV;
        Vt[bh * DV * NW + dv * NW + key] = (_Float16)V[i];
    }
    // mask -> both tile-major bias layouts
    {
        const float bv = (mask[i] == 0) ? -1.0e9f : 0.0f;
        const int q = (int)(i / NW), k = (int)(i % NW);
        const int qt = q >> 4, nq = q & 15;
        // pass-1 (S^T): C lane = h*16 + (q%16), reg = key%8, tile = (key%32)/16
        {
            const int kb = k >> 5, kk = k & 31;
            const int tile = kk >> 4, rr = kk & 15;
            const int lane = (rr >> 3) * 16 + nq;
            const int j = tile * 8 + (rr & 7);
            biasT[(((size_t)qt * KB32 + kb) << 9) + lane * 16 + j] = bv;
        }
        // pass-2 (S): C lane = h*16 + (k%16), reg = q%8
        {
            const int kt = k >> 4, nk = k & 15;
            const int lane = ((nq >> 3) * 16) + nk;
            biasR[(((size_t)qt * QT + kt) << 8) + lane * 8 + (nq & 7)] = bv;
        }
    }
}

// ---------------- main fused attention ----------------
__global__ __launch_bounds__(256) void attn_wmma_kernel(
    const _Float16* __restrict__ Qh, const _Float16* __restrict__ Kh,
    const _Float16* __restrict__ Vt, const float* __restrict__ biasT,
    const float* __restrict__ biasR,
    float* __restrict__ attn_out, float* __restrict__ probs_out)
{
    __shared__ __align__(16) float o_stage[WAVES_PER_BLOCK][16 * OSTRIDE];

    const int lane = threadIdx.x & 31;
    const int widx = threadIdx.x >> 5;
    const int gw   = blockIdx.x * WAVES_PER_BLOCK + widx;
    const int bh   = gw / QT;
    const int qt   = gw % QT;
    const int q0   = qt * 16;
    const int n    = lane & 15;    // C/B column, A row
    const int h    = lane >> 4;    // lane half

    const _Float16* Qb = Qh + (size_t)bh * NW * DK;
    const _Float16* Kb = Kh + (size_t)bh * NW * DK;
    const _Float16* Vb = Vt + (size_t)bh * DV * NW;   // [dv][key]

    // Q tile (two DK=32 chunks). Identical lane mapping serves as:
    //   pass 1: B operand of S^T = K x Q^T  (col = query)
    //   pass 2: A operand of S   = Q x K^T  (row = query)
    const _Float16* qrow = Qb + (size_t)(q0 + n) * DK;
    const v16h bq0 = load_ab(qrow, h);
    const v16h bq1 = load_ab(qrow + 32, h);

    // O^T accumulators: rows = dv (r + 8h), cols = query (lane n)
    v8f O0 = {}, O1 = {}, O2 = {}, O3 = {};
    float mrun = -3.0e38f, lrun = 0.0f;   // per-lane: this lane's query column

    // ======== pass 1: S^T = K·Q^T (+bias C) -> per-lane softmax -> O^T += V^T·P^T ========
    for (int kb = 0; kb < KB32; ++kb) {
        const int k0 = kb * 32;
        if (kb + 1 < KB32)
            __builtin_prefetch(Kb + (size_t)(k0 + 32 + n) * DK, 0, 0);

        // K tiles as A operands (row = key)
        const _Float16* kr0 = Kb + (size_t)(k0 + n) * DK;
        const _Float16* kr1 = Kb + (size_t)(k0 + 16 + n) * DK;
        const v16h ak00 = load_ab(kr0, h), ak01 = load_ab(kr0 + 32, h);
        const v16h ak10 = load_ab(kr1, h), ak11 = load_ab(kr1 + 32, h);

        // bias as initial accumulators: 16 contiguous floats per lane (4x b128)
        const float* bt = biasT + (((size_t)qt * KB32 + kb) << 9) + lane * 16;
        v8f s0, s1;
        #pragma unroll
        for (int r = 0; r < 8; ++r) { s0[r] = bt[r]; s1[r] = bt[8 + r]; }

        s0 = __builtin_amdgcn_wmma_f32_16x16x32_f16(false, ak00, false, bq0, (short)0, s0, false, false);
        s0 = __builtin_amdgcn_wmma_f32_16x16x32_f16(false, ak01, false, bq1, (short)0, s0, false, false);
        s1 = __builtin_amdgcn_wmma_f32_16x16x32_f16(false, ak10, false, bq0, (short)0, s1, false, false);
        s1 = __builtin_amdgcn_wmma_f32_16x16x32_f16(false, ak11, false, bq1, (short)0, s1, false, false);

        // per-lane online softmax over this block's 32 keys (16 in-lane + half-merge)
        float tm = fmaxf(tree_max8(s0), tree_max8(s1));
        tm = fmaxf(tm, __shfl_xor(tm, 16, 32));
        const float mnew = fmaxf(mrun, tm);
        const float nm2  = -mnew * L2E;
        const float alpha = __builtin_amdgcn_exp2f(fmaf(mrun, L2E, nm2));
        mrun = mnew;
        #pragma unroll
        for (int r = 0; r < 8; ++r) {
            s0[r] = exp_fma(s0[r], nm2);
            s1[r] = exp_fma(s1[r], nm2);
        }
        float rs = tree_sum8(s0) + tree_sum8(s1);
        rs += __shfl_xor(rs, 16, 32);
        lrun = lrun * alpha + rs;

        #pragma unroll
        for (int r = 0; r < 8; ++r) {
            O0[r] *= alpha; O1[r] *= alpha; O2[r] *= alpha; O3[r] *= alpha;
        }

        // P^T is already in B-operand layout: e<8 -> tile0 reg e, e>=8 -> tile1 reg e-8
        v16h bp;
        #pragma unroll
        for (int e = 0; e < 8; ++e) {
            bp[e]     = (_Float16)s0[e];
            bp[e + 8] = (_Float16)s1[e];
        }

        // O^T[c] += V^T(A) x P^T(B); V^T rows = dv, contiguous runs from Vt
        #pragma unroll
        for (int c = 0; c < 4; ++c) {
            const v16h av = load_ab(Vb + (size_t)(c * 16 + n) * NW + k0, h);
            v8f acc = (c == 0) ? O0 : (c == 1) ? O1 : (c == 2) ? O2 : O3;
            acc = __builtin_amdgcn_wmma_f32_16x16x32_f16(false, av, false, bp, (short)0, acc, false, false);
            if (c == 0) O0 = acc; else if (c == 1) O1 = acc; else if (c == 2) O2 = acc; else O3 = acc;
        }
    }

    const float linv = 1.0f / lrun;   // per-lane (per-query) normalizer

    // ---- epilogue: normalize (lane-local!), transpose O^T via LDS, coalesced b128 stores ----
    #pragma unroll
    for (int r = 0; r < 8; ++r) {
        o_stage[widx][n * OSTRIDE + (0 * 16 + r + 8 * h)] = O0[r] * linv;
        o_stage[widx][n * OSTRIDE + (1 * 16 + r + 8 * h)] = O1[r] * linv;
        o_stage[widx][n * OSTRIDE + (2 * 16 + r + 8 * h)] = O2[r] * linv;
        o_stage[widx][n * OSTRIDE + (3 * 16 + r + 8 * h)] = O3[r] * linv;
    }
    asm volatile("s_wait_dscnt 0" ::: "memory");
    {
        float* orow = attn_out + ((size_t)bh * NW + q0 + n) * DV + h * 32;
        #pragma unroll
        for (int j = 0; j < 8; ++j) {
            v4f t = *(const v4f*)&o_stage[widx][n * OSTRIDE + h * 32 + 4 * j];
            *(v4f*)(orow + 4 * j) = t;
        }
    }

    // hand per-query stats from lane-columns to C-rows for pass 2
    float nm2row[8], lirow[8];
    #pragma unroll
    for (int r = 0; r < 8; ++r) {
        nm2row[r] = __shfl(-mrun * L2E, r + 8 * h, 32);
        lirow[r]  = __shfl(linv, r + 8 * h, 32);
    }

    // ======== pass 2: S = Q·K^T (+bias C), write normalized probs (row-contiguous) ========
    for (int kt = 0; kt < QT; ++kt) {
        const int k0 = kt * 16;
        const _Float16* kr = Kb + (size_t)(k0 + n) * DK;
        const v16h b0 = load_ab(kr, h);
        const v16h b1 = load_ab(kr + 32, h);

        const float* br = biasR + (((size_t)qt * QT + kt) << 8) + lane * 8;
        v8f s;
        #pragma unroll
        for (int r = 0; r < 8; ++r) s[r] = br[r];

        s = __builtin_amdgcn_wmma_f32_16x16x32_f16(false, bq0, false, b0, (short)0, s, false, false);
        s = __builtin_amdgcn_wmma_f32_16x16x32_f16(false, bq1, false, b1, (short)0, s, false, false);

        #pragma unroll
        for (int r = 0; r < 8; ++r) {
            const int q = q0 + r + 8 * h;
            const float p = exp_fma(s[r], nm2row[r]) * lirow[r];
            probs_out[((size_t)bh * NW + q) * NW + k0 + n] = p;
        }
    }
}

extern "C" void kernel_launch(void* const* d_in, const int* in_sizes, int n_in,
                              void* d_out, int out_size, void* d_ws, size_t ws_size,
                              hipStream_t stream) {
    (void)in_sizes; (void)n_in; (void)ws_size; (void)out_size;
    const float* Q    = (const float*)d_in[0];
    const float* K    = (const float*)d_in[1];
    const float* V    = (const float*)d_in[2];
    const int*   mask = (const int*)d_in[3];

    // ws layout: Qh | Kh | Vt (f16, NQ each) | biasT | biasR (f32, NW*NW each) ~= 59 MB
    _Float16* Qh    = (_Float16*)d_ws;
    _Float16* Kh    = Qh + NQ;
    _Float16* Vt    = Kh + NQ;
    float*    biasT = (float*)(Vt + NQ);
    float*    biasR = biasT + NQ;

    float* attn  = (float*)d_out;
    float* probs = attn + (size_t)NBH * NW * DV;   // outputs concatenated: (attn, probs)

    prep_kernel<<<(unsigned)((NQ + 255) / 256), 256, 0, stream>>>(
        Q, K, V, mask, Qh, Kh, Vt, biasT, biasR);

    const int total_waves = NBH * QT;              // 4096 query tiles
    dim3 grid(total_waves / WAVES_PER_BLOCK);      // 512 blocks
    dim3 block(32 * WAVES_PER_BLOCK);              // 8 waves / block (wave32)
    attn_wmma_kernel<<<grid, block, 0, stream>>>(Qh, Kh, Vt, biasT, biasR, attn, probs);
}
